// ConvBertSDConv_24558622999248
// MI455X (gfx1250) — matrix operand order repaired and checked by reading, not compiled, not run on hardware
//
#include <hip/hip_runtime.h>
#include <hip/hip_bf16.h>

// ---------------------------------------------------------------------------
// ConvBert SDConv fused pipeline for gfx1250 (MI455X), wave32 + WMMA + TDM.
// B=8, S=2048, C=HID=768, H=12, K=9, D=64, M=B*S=16384.
// ---------------------------------------------------------------------------

typedef __attribute__((ext_vector_type(16))) __bf16 v16bf;
typedef __attribute__((ext_vector_type(8)))  __bf16 v8bf;
typedef __attribute__((ext_vector_type(8)))  float  v8f;
typedef __attribute__((ext_vector_type(4)))  unsigned int u32x4;
typedef __attribute__((ext_vector_type(8)))  int i32x8;
typedef __attribute__((ext_vector_type(4)))  int i32x4;

#define M_TOK   16384
#define SEQ     2048
#define CH      768
#define HEADS   12
#define KSZ     9
#define PADW    4
#define NPAD    112      // H*K = 108 padded to 7 tiles of 16

// ---- WMMA fragment loaders (row-major source [rows, stride] of bf16) ------
// A-matrix 16x32 bf16 layout (ISA 7.12.2): lanes 0-15 hold row M=lane,
// K = {0..7, 16..23}; lanes 16-31 hold row M=lane-16, K = {8..15, 24..31}.
__device__ inline v16bf load_fragA(const __bf16* base, int row0, int stride,
                                   int k0, int lane) {
  const __bf16* p = base + (size_t)(row0 + (lane & 15)) * stride
                         + k0 + ((lane >> 4) << 3);
  v8bf lo = *reinterpret_cast<const v8bf*>(p);        // K kb..kb+7
  v8bf hi = *reinterpret_cast<const v8bf*>(p + 16);   // K kb+16..kb+23
  v16bf out;
#pragma unroll
  for (int i = 0; i < 8; ++i) { out[i] = lo[i]; out[i + 8] = hi[i]; }
  return out;
}

// B-matrix 32x16 bf16: lanes 0-15 hold col N=lane, K={0..7,16..23}; lanes 16-31
// hold col N=lane-16, K={8..15,24..31}. Source stored [N, K] row-major.
__device__ inline v16bf load_fragB(const __bf16* base, int row0, int stride,
                                   int k0, int lane) {
  const __bf16* p = base + (size_t)(row0 + (lane & 15)) * stride
                         + k0 + ((lane >> 4) << 3);
  v8bf lo = *reinterpret_cast<const v8bf*>(p);        // K kb..kb+7
  v8bf hi = *reinterpret_cast<const v8bf*>(p + 16);   // K kb+16..kb+23
  v16bf out;
#pragma unroll
  for (int i = 0; i < 8; ++i) { out[i] = lo[i]; out[i + 8] = hi[i]; }
  return out;
}

__device__ inline void softmax9(float* f) {
  float mx = f[0];
#pragma unroll
  for (int k = 1; k < KSZ; ++k) mx = fmaxf(mx, f[k]);
  float s = 0.f;
#pragma unroll
  for (int k = 0; k < KSZ; ++k) { f[k] = __expf(f[k] - mx); s += f[k]; }
  float inv = 1.f / s;
#pragma unroll
  for (int k = 0; k < KSZ; ++k) f[k] *= inv;
}

// ---------------------------------------------------------------------------
// Kernel 1: depthwise conv over sequence, fp32 in -> bf16 out [M, 768]
// ---------------------------------------------------------------------------
__global__ __launch_bounds__(256) void dw_conv_kernel(
    const float* __restrict__ hidden,   // [B, S, HID]
    const float* __restrict__ dww,      // [HID, 1, K] -> [c][k]
    __bf16* __restrict__ dwA)           // [M, HID] bf16
{
  int idx = blockIdx.x * 256 + threadIdx.x;       // < M*CH
  int c = idx % CH;
  int m = idx / CH;
  int b = m >> 11;                                 // /2048
  int s = m & (SEQ - 1);
  float acc = 0.f;
#pragma unroll
  for (int k = 0; k < KSZ; ++k) {
    int sp = s + k - PADW;
    if (sp >= 0 && sp < SEQ)
      acc += hidden[((size_t)b * SEQ + sp) * CH + c] * dww[c * KSZ + k];
  }
  dwA[idx] = (__bf16)acc;
}

// ---------------------------------------------------------------------------
// Kernel 2: weight conversion to bf16 (+ zero-pad attn_W rows 108..111)
// ---------------------------------------------------------------------------
__global__ __launch_bounds__(256) void cvt_weights_kernel(
    const float* __restrict__ pw,       // [C, HID]
    const float* __restrict__ attnW,    // [108, C]
    __bf16* __restrict__ pwB,           // [C, HID] bf16
    __bf16* __restrict__ attnB)         // [112, C] bf16
{
  int idx = blockIdx.x * 256 + threadIdx.x;
  const int NPW = CH * CH;                          // 589824
  if (idx < NPW) {
    pwB[idx] = (__bf16)pw[idx];
  } else {
    int j = idx - NPW;                              // < 112*768
    int row = j / CH;
    attnB[j] = (row < HEADS * KSZ) ? (__bf16)attnW[j] : (__bf16)0.f;
  }
}

// ---------------------------------------------------------------------------
// Kernel 3: TDM(A tile -> LDS) -> GEMM1(WMMA) -> gate -> GEMM2(WMMA)
//           -> softmax^2 -> lightconv.  One block = one 16-token M-tile.
// ---------------------------------------------------------------------------
__global__ __launch_bounds__(256) void fused_kernel(
    const __bf16* __restrict__ dwA,      // [M, 768] bf16
    const __bf16* __restrict__ pwB,      // [768, 768] bf16
    const __bf16* __restrict__ attnB,    // [112, 768] bf16
    const float* __restrict__ query,     // [M, 768]
    const float* __restrict__ value,     // [B, S, 768]
    const float* __restrict__ sep_bias,  // [768]
    const float* __restrict__ attn_b,    // [108]
    float* __restrict__ out)             // [M, 768] = [B,S,H,D]
{
  __shared__ __attribute__((aligned(16))) __bf16 ldsIn[16 * CH];  // dwA tile via TDM (24 KB)
  __shared__ __attribute__((aligned(16))) __bf16 ldsA[16 * CH];   // conv_attn tile (24 KB)
  __shared__ __attribute__((aligned(16))) float  ldsF[16 * NPAD]; // filt tile (7 KB)

  const int tid  = threadIdx.x;
  const int lane = tid & 31;
  const int w    = tid >> 5;
  const int m0   = blockIdx.x * 16;

  // ====== Stage 0: DMA the A tile [16 x 768] bf16 into LDS (TDM) ===========
#if __has_builtin(__builtin_amdgcn_tensor_load_to_lds)
  if (w == 0) {
    const unsigned long long ga =
        (unsigned long long)(uintptr_t)(dwA + (size_t)m0 * CH);
    const unsigned lds_addr = (unsigned)(uintptr_t)(&ldsIn[0]); // LDS aperture low bits

    u32x4 g0;
    g0[0] = 1u;                                    // count=1, user descriptor
    g0[1] = lds_addr;                              // lds_addr (bytes)
    g0[2] = (unsigned)(ga & 0xFFFFFFFFu);          // global_addr[31:0]
    g0[3] = ((unsigned)(ga >> 32) & 0x01FFFFFFu)   // global_addr[56:32]
            | (2u << 30);                          // type = 2 ("image")

    i32x8 g1;
    g1[0] = 0x00010000;                 // wg_mask=0, data_size=1 (2 bytes)
    g1[1] = (int)(768u << 16);          // tensor_dim0 = 768 (bits 79:48 lo)
    g1[2] = (int)((unsigned)M_TOK << 16); // tensor_dim0 hi | tensor_dim1 lo
    g1[3] = (int)(768u << 16);          // tensor_dim1 hi | tile_dim0 = 768
    g1[4] = 16;                         // tile_dim1 = 16, tile_dim2 = 0
    g1[5] = 768;                        // tensor_dim0_stride = 768
    g1[6] = 0;                          // stride hi | tensor_dim1_stride lo
    g1[7] = 0;

    i32x4 gz = {0, 0, 0, 0};            // groups 2/3: dims 3/4 unused
#if defined(__clang_major__) && (__clang_major__ >= 23)
    i32x8 gz8 = {0, 0, 0, 0, 0, 0, 0, 0};
    __builtin_amdgcn_tensor_load_to_lds(g0, g1, gz, gz, gz8, 0);
#else
    __builtin_amdgcn_tensor_load_to_lds(g0, g1, gz, gz, 0);
#endif
    __builtin_amdgcn_s_wait_tensorcnt(0);
  }
#else
  // Fallback: cooperative copy, 6144 dwords / 256 threads = 24 each.
  {
    const unsigned* src = (const unsigned*)(dwA + (size_t)m0 * CH);
    unsigned* dst = (unsigned*)&ldsIn[0];
    for (int i = tid; i < 16 * CH / 2; i += 256) dst[i] = src[i];
  }
#endif
  __syncthreads();

  // ================= GEMM1: ks_out tile, 8 waves x 6 N-tiles ================
  const v8f vzero = {0.f, 0.f, 0.f, 0.f, 0.f, 0.f, 0.f, 0.f};
  v8f acc[6] = {vzero, vzero, vzero, vzero, vzero, vzero};
  const int n0w = w * 96;

  for (int kk = 0; kk < 24; ++kk) {
    const int k0 = kk * 32;
    v16bf a = load_fragA(ldsIn, 0, CH, k0, lane);            // ds_load path
    if (kk < 23)  // stream-prefetch next B slice (global_prefetch_b8)
      __builtin_prefetch(pwB + (size_t)(n0w + (lane & 15)) * CH + k0 + 32, 0, 1);
#pragma unroll
    for (int t = 0; t < 6; ++t) {
      v16bf b = load_fragB(pwB, n0w + t * 16, CH, k0, lane);
      acc[t] = __builtin_amdgcn_wmma_f32_16x16x32_bf16(
          false, a, false, b, (short)0, acc[t], false, false);
    }
  }

  // Epilogue: + sep_bias, * query  -> bf16 conv_attn tile in LDS.
  // C layout: VGPR r -> M = r + 8*(lane>=16), N = lane&15.
  {
    const int mhalf = (lane >> 4) << 3;
#pragma unroll
    for (int t = 0; t < 6; ++t) {
      const int n = n0w + t * 16 + (lane & 15);
      const float sb = sep_bias[n];
#pragma unroll
      for (int r = 0; r < 8; ++r) {
        const int mloc = mhalf + r;
        float v = acc[t][r] + sb;
        v *= query[(size_t)(m0 + mloc) * CH + n];
        ldsA[mloc * CH + n] = (__bf16)v;
      }
    }
  }
  __syncthreads();

  // ============ GEMM2: filt = conv_attn @ attn_W^T, waves 0..6 ==============
  if (w < 7) {
    v8f facc = vzero;
    for (int kk = 0; kk < 24; ++kk) {
      const int k0 = kk * 32;
      v16bf a = load_fragA(ldsA, 0, CH, k0, lane);           // ds_load path
      v16bf b = load_fragB(attnB, w * 16, CH, k0, lane);
      facc = __builtin_amdgcn_wmma_f32_16x16x32_bf16(
          false, a, false, b, (short)0, facc, false, false);
    }
    const int n = w * 16 + (lane & 15);
    const float bb = (n < HEADS * KSZ) ? attn_b[n] : 0.f;
    const int mhalf = (lane >> 4) << 3;
#pragma unroll
    for (int r = 0; r < 8; ++r)
      ldsF[(mhalf + r) * NPAD + n] = facc[r] + bb;
  }
  __syncthreads();

  // ============ double softmax over K + light conv, 192 threads =============
  if (tid < 16 * HEADS) {
    const int row = tid / HEADS;                 // 0..15 token in tile
    const int h   = tid % HEADS;
    float f[KSZ];
#pragma unroll
    for (int k = 0; k < KSZ; ++k) f[k] = ldsF[row * NPAD + h * KSZ + k];
    softmax9(f);
    softmax9(f);

    const int b = m0 >> 11;
    const int s = (m0 & (SEQ - 1)) + row;
    const float* vbase = value + (size_t)b * SEQ * CH + h * 64;
    float* obase = out + (size_t)(m0 + row) * CH + h * 64;
#pragma unroll 4
    for (int d = 0; d < 64; ++d) {
      float a = 0.f;
#pragma unroll
      for (int k = 0; k < KSZ; ++k) {
        const int sp = s + k - PADW;
        if (sp >= 0 && sp < SEQ)
          a += vbase[(size_t)sp * CH + d] * f[k];
      }
      obase[d] = a;
    }
  }
}

// ---------------------------------------------------------------------------
extern "C" void kernel_launch(void* const* d_in, const int* in_sizes, int n_in,
                              void* d_out, int out_size, void* d_ws, size_t ws_size,
                              hipStream_t stream) {
  const float* query    = (const float*)d_in[0];   // [8,2048,768]
  const float* value    = (const float*)d_in[1];   // [8,2048,768]
  const float* hidden   = (const float*)d_in[2];   // [8,2048,768]
  const float* dw_w     = (const float*)d_in[3];   // [768,1,9]
  const float* pw_w     = (const float*)d_in[4];   // [768,768,1]
  const float* sep_bias = (const float*)d_in[5];   // [768]
  const float* attn_W   = (const float*)d_in[6];   // [108,768]
  const float* attn_b   = (const float*)d_in[7];   // [108]
  float* out = (float*)d_out;

  // Workspace layout (16B aligned chunks):
  char* ws = (char*)d_ws;
  __bf16* dwA   = (__bf16*)(ws);                                  // 16384*768*2 = 25165824
  __bf16* pwB   = (__bf16*)(ws + 25165824);                       // 768*768*2   = 1179648
  __bf16* attnB = (__bf16*)(ws + 25165824 + 1179648);             // 112*768*2   = 172032
  (void)in_sizes; (void)n_in; (void)out_size; (void)ws_size;

  // 1) depthwise conv -> bf16 A matrix
  dw_conv_kernel<<<(M_TOK * CH) / 256, 256, 0, stream>>>(hidden, dw_w, dwA);

  // 2) weight conversions (768*768 + 112*768 = 675840 elems)
  cvt_weights_kernel<<<675840 / 256, 256, 0, stream>>>(pw_w, attn_W, pwB, attnB);

  // 3) fused WMMA pipeline: one block per 16 tokens
  fused_kernel<<<M_TOK / 16, 256, 0, stream>>>(
      dwA, pwB, attnB, query, value, sep_bias, attn_b, out);
}
